// MambaBlock_6184752906481
// MI455X (gfx1250) — compile-verified, matrix operand order checked by reading, not run on hardware
//
#include <hip/hip_runtime.h>
#include <cstdint>
#include <cstddef>

// ---------------------------------------------------------------------------
// Mamba block for MI455X (gfx1250, wave32, WMMA).
// GEMMs run in bf16 via v_wmma_f32_16x16x32_bf16 with f32 accumulation.
// LDS staging uses gfx1250 async global->LDS copies (ASYNCcnt) when the
// toolchain exposes the builtin, with a synchronous fallback.
// ---------------------------------------------------------------------------

typedef __attribute__((ext_vector_type(16))) __bf16        v16bf;
typedef __attribute__((ext_vector_type(8)))  float         v8f;
typedef __attribute__((ext_vector_type(4)))  unsigned int  u32x4;
typedef __attribute__((ext_vector_type(4)))  int           v4i;

typedef __attribute__((address_space(1))) v4i as1_v4i;   // global int4
typedef __attribute__((address_space(3))) v4i as3_v4i;   // LDS int4

#if defined(__has_builtin)
#  if __has_builtin(__builtin_amdgcn_global_load_async_to_lds_b128)
#    define USE_ASYNC_LDS 1
#  else
#    define USE_ASYNC_LDS 0
#  endif
#else
#  define USE_ASYNC_LDS 0
#endif

#define D_MODEL 768
#define D_INNER 1536
#define D_XZ    3072
#define D_STATE 16
#define D_CONVK 4
#define DT_RANK 48
#define D_DBL   80          // DT_RANK + 2*D_STATE
#define D_FFN   3072
#define BATCH   4
#define SEQ     2048
#define NTOK    (BATCH * SEQ)   // 8192

static __device__ __forceinline__ unsigned short f2bf(float f) {
    unsigned int u = __float_as_uint(f);
    unsigned int r = (u + 0x7FFFu + ((u >> 16) & 1u)) >> 16;  // RNE
    return (unsigned short)r;
}

// 16-byte global -> LDS copy: async (ASYNCcnt) on gfx1250 toolchains that
// expose the builtin, otherwise through VGPRs.
static __device__ __forceinline__ void copy16_g2l(const unsigned short* g,
                                                  unsigned short* l) {
#if USE_ASYNC_LDS
    __builtin_amdgcn_global_load_async_to_lds_b128(
        (as1_v4i*)(v4i*)(void*)g, (as3_v4i*)(v4i*)(void*)l, 0, 0);
#else
    *reinterpret_cast<u32x4*>(l) = *reinterpret_cast<const u32x4*>(g);
#endif
}

static __device__ __forceinline__ void wait_async_copies() {
#if USE_ASYNC_LDS
#  if defined(__has_builtin)
#    if __has_builtin(__builtin_amdgcn_s_wait_asynccnt)
    __builtin_amdgcn_s_wait_asynccnt(0);
#    else
    asm volatile("s_wait_asynccnt 0x0" ::: "memory");
#    endif
#  else
    asm volatile("s_wait_asynccnt 0x0" ::: "memory");
#  endif
#endif
}

// ---------------------------------------------------------------------------
// f32 -> bf16 conversion
// ---------------------------------------------------------------------------
__global__ void cvt_f32_bf16_kernel(const float* __restrict__ src,
                                    unsigned short* __restrict__ dst, int n) {
    int i = blockIdx.x * blockDim.x + threadIdx.x;
    if (i < n) dst[i] = f2bf(src[i]);
}

// ---------------------------------------------------------------------------
// C[M,N] = A[M,K] (bf16, row stride lda) * W[N,K]^T (bf16, row stride ldw)
// optionally + bias[N].  Block tile 128x128, 8 waves, each wave 64x32 via
// 4x2 fragments of v_wmma_f32_16x16x32_bf16.  K staged in 32-wide LDS tiles,
// double-buffered; full tiles stream via async global->LDS copies, edge
// tiles (K=48 tail, N=80 block) take a zero-filled synchronous path.
// ---------------------------------------------------------------------------
__global__ __launch_bounds__(256)
void gemm_bf16_wmma_kernel(const unsigned short* __restrict__ A, int lda,
                           const unsigned short* __restrict__ W, int ldw,
                           const float* __restrict__ bias,
                           float* __restrict__ C, int ldc,
                           int M, int N, int K) {
    constexpr int TS = 40;                     // LDS row stride (halves), 80B
    __shared__ unsigned short As[2][128 * TS]; // 2 x 10 KB
    __shared__ unsigned short Bs[2][128 * TS]; // 2 x 10 KB

    const int tid  = threadIdx.x;
    const int lane = tid & 31;
    const int wave = tid >> 5;
    const int wm   = wave & 1;                 // 2 M-slabs of 64 rows
    const int wn   = wave >> 1;                // 4 N-slabs of 32 cols
    const int m0   = blockIdx.y * 128;
    const int n0   = blockIdx.x * 128;

    const int fr = lane & 15;                  // fragment row within 16
    const int ko = (lane >> 4) << 3;           // K half-offset: 0 or 8

    v8f acc[4][2];
    {
        v8f z;
        #pragma unroll
        for (int r = 0; r < 8; ++r) z[r] = 0.0f;
        #pragma unroll
        for (int i = 0; i < 4; ++i)
            #pragma unroll
            for (int j = 0; j < 2; ++j) acc[i][j] = z;
    }

    union FragU { v16bf v; u32x4 q[2]; };

    // Stage one 128x32 A tile + 128x32 W tile into LDS buffer `buf`.
    auto stage = [&](int kt, int buf) {
        const int  kbase = kt * 32;
        const bool fullK = (kbase + 32 <= K);
        const bool fastA = fullK && (m0 + 128 <= M);
        const bool fastB = fullK && (n0 + 128 <= N);
        #pragma unroll
        for (int t = 0; t < 2; ++t) {
            int idx = tid + t * 256;           // 512 16B chunks per tile
            int row = idx >> 2;
            int c8  = (idx & 3) << 3;          // 0,8,16,24 halves
            int k   = kbase + c8;
            unsigned short* lA = &As[buf][row * TS + c8];
            unsigned short* lB = &Bs[buf][row * TS + c8];
            if (fastA) {
                copy16_g2l(A + (size_t)(m0 + row) * lda + k, lA);
            } else {
                u32x4 va = {0u, 0u, 0u, 0u};
                int gm = m0 + row;
                if (gm < M && (k + 8) <= K)
                    va = *reinterpret_cast<const u32x4*>(A + (size_t)gm * lda + k);
                *reinterpret_cast<u32x4*>(lA) = va;
            }
            if (fastB) {
                copy16_g2l(W + (size_t)(n0 + row) * ldw + k, lB);
            } else {
                u32x4 vb = {0u, 0u, 0u, 0u};
                int gn = n0 + row;
                if (gn < N && (k + 8) <= K)
                    vb = *reinterpret_cast<const u32x4*>(W + (size_t)gn * ldw + k);
                *reinterpret_cast<u32x4*>(lB) = vb;
            }
        }
    };

    const int kTiles = (K + 31) / 32;
    stage(0, 0);
    for (int kt = 0; kt < kTiles; ++kt) {
        const int buf = kt & 1;
        wait_async_copies();       // this wave's fills of `buf` complete
        __syncthreads();           // all waves' fills of `buf` complete
        if (kt + 1 < kTiles) stage(kt + 1, buf ^ 1);  // overlap with WMMA

        // Fragment loads per documented 16-bit A layout:
        // lane<16 holds K[ko..ko+7] and K[ko+16..ko+23] (two 16B chunks).
        FragU a[4], b[2];
        #pragma unroll
        for (int i = 0; i < 4; ++i) {
            const unsigned short* p = &As[buf][(wm * 64 + i * 16 + fr) * TS + ko];
            a[i].q[0] = *reinterpret_cast<const u32x4*>(p);
            a[i].q[1] = *reinterpret_cast<const u32x4*>(p + 16);
        }
        #pragma unroll
        for (int j = 0; j < 2; ++j) {
            const unsigned short* p = &Bs[buf][(wn * 32 + j * 16 + fr) * TS + ko];
            b[j].q[0] = *reinterpret_cast<const u32x4*>(p);
            b[j].q[1] = *reinterpret_cast<const u32x4*>(p + 16);
        }
        #pragma unroll
        for (int i = 0; i < 4; ++i)
            #pragma unroll
            for (int j = 0; j < 2; ++j)
                acc[i][j] = __builtin_amdgcn_wmma_f32_16x16x32_bf16(
                    false, a[i].v, false, b[j].v, (short)0, acc[i][j],
                    false, false);
    }

    // Store: C/D layout — VGPR r: lanes 0-15 -> M=r, lanes 16-31 -> M=8+r.
    const int mo = (lane < 16) ? 0 : 8;
    #pragma unroll
    for (int i = 0; i < 4; ++i) {
        #pragma unroll
        for (int j = 0; j < 2; ++j) {
            int gn = n0 + wn * 32 + j * 16 + fr;
            if (gn >= N) continue;
            float bv = bias ? bias[gn] : 0.0f;
            #pragma unroll
            for (int r = 0; r < 8; ++r) {
                int gm = m0 + wm * 64 + i * 16 + mo + r;
                if (gm < M) C[(size_t)gm * ldc + gn] = acc[i][j][r] + bv;
            }
        }
    }
}

// ---------------------------------------------------------------------------
// Causal depthwise conv1d (kernel 4) + SiLU over xc = xz[..., :D_INNER].
// Writes f32 and bf16 copies.
// ---------------------------------------------------------------------------
__global__ void conv_silu_kernel(const float* __restrict__ xz,
                                 const float* __restrict__ conv_w,
                                 const float* __restrict__ conv_b,
                                 float* __restrict__ xconv,
                                 unsigned short* __restrict__ xconv_bf) {
    int i = blockIdx.x * blockDim.x + threadIdx.x;
    if (i >= NTOK * D_INNER) return;
    int d = i % D_INNER;
    int t = i / D_INNER;           // token = b*SEQ + l
    int l = t % SEQ;
    float acc = conv_b[d];
    #pragma unroll
    for (int k = 0; k < D_CONVK; ++k) {
        int ll = l + k - (D_CONVK - 1);
        if (ll >= 0)
            acc += conv_w[d * D_CONVK + k] *
                   xz[(size_t)(t + ll - l) * D_XZ + d];
    }
    float s = acc / (1.0f + expf(-acc));       // SiLU
    xconv[i]    = s;
    xconv_bf[i] = f2bf(s);
}

// ---------------------------------------------------------------------------
// dt = softplus(dt + dt_proj_b)
// ---------------------------------------------------------------------------
__global__ void dt_softplus_kernel(float* __restrict__ dt,
                                   const float* __restrict__ dtb) {
    int i = blockIdx.x * blockDim.x + threadIdx.x;
    if (i >= NTOK * D_INNER) return;
    int d = i % D_INNER;
    float v = dt[i] + dtb[d];
    dt[i] = (v > 20.0f) ? v : log1pf(expf(v));
}

// ---------------------------------------------------------------------------
// Selective scan.  16 lanes per channel (state dim across lanes), wave32.
// y = (sum_s h_s*C_s + Dskip*x) * silu(z), stored as bf16 for out_proj.
// ---------------------------------------------------------------------------
__global__ __launch_bounds__(256)
void scan_kernel(const float* __restrict__ dt, const float* __restrict__ xconv,
                 const float* __restrict__ xdbl, const float* __restrict__ xz,
                 const float* __restrict__ A_log, const float* __restrict__ Dskip,
                 unsigned short* __restrict__ y_bf) {
    int tid = threadIdx.x;
    int s   = tid & 15;
    int ch  = blockIdx.x * 16 + (tid >> 4);
    if (ch >= BATCH * D_INNER) return;
    int b = ch / D_INNER;
    int d = ch % D_INNER;
    float A  = -expf(A_log[d * D_STATE + s]);
    float Dv = Dskip[d];
    float h  = 0.0f;
    size_t tokBase = (size_t)b * SEQ;
    for (int l = 0; l < SEQ; ++l) {
        size_t t  = tokBase + l;
        float dtv = dt[t * D_INNER + d];
        float xv  = xconv[t * D_INNER + d];
        float Bv  = xdbl[t * D_DBL + DT_RANK + s];
        float Cv  = xdbl[t * D_DBL + DT_RANK + D_STATE + s];
        h = expf(dtv * A) * h + dtv * xv * Bv;
        float p = h * Cv;
        p += __shfl_xor(p, 1, 32);
        p += __shfl_xor(p, 2, 32);
        p += __shfl_xor(p, 4, 32);
        p += __shfl_xor(p, 8, 32);
        if (s == 0) {
            float z = xz[t * D_XZ + D_INNER + d];
            float y = (p + Dv * xv) * (z / (1.0f + expf(-z)));
            y_bf[t * D_INNER + d] = f2bf(y);
        }
    }
}

// ---------------------------------------------------------------------------
// out = LayerNorm(xin + alpha*res) * g + b ; optional bf16 copy of out.
// One 256-thread block per row of 768.
// ---------------------------------------------------------------------------
__global__ __launch_bounds__(256)
void res_ln_kernel(const float* __restrict__ xin, const float* __restrict__ res,
                   const float* __restrict__ alpha,
                   const float* __restrict__ g, const float* __restrict__ bta,
                   float* __restrict__ out, unsigned short* __restrict__ out_bf) {
    __shared__ float vbuf[D_MODEL];
    __shared__ float s1[256];
    __shared__ float s2[256];
    int tid = threadIdx.x;
    size_t row = blockIdx.x;
    float a = alpha[0];
    float p1 = 0.0f, p2 = 0.0f;
    #pragma unroll
    for (int i = 0; i < 3; ++i) {
        int idx = i * 256 + tid;
        float v = xin[row * D_MODEL + idx] + a * res[row * D_MODEL + idx];
        vbuf[idx] = v;
        p1 += v;
        p2 += v * v;
    }
    s1[tid] = p1;
    s2[tid] = p2;
    __syncthreads();
    for (int st = 128; st > 0; st >>= 1) {
        if (tid < st) { s1[tid] += s1[tid + st]; s2[tid] += s2[tid + st]; }
        __syncthreads();
    }
    float mean = s1[0] * (1.0f / D_MODEL);
    float var  = s2[0] * (1.0f / D_MODEL) - mean * mean;
    float rstd = rsqrtf(var + 1e-5f);
    #pragma unroll
    for (int i = 0; i < 3; ++i) {
        int idx = i * 256 + tid;
        float o = g[idx] * (vbuf[idx] - mean) * rstd + bta[idx];
        out[row * D_MODEL + idx] = o;
        if (out_bf) out_bf[row * D_MODEL + idx] = f2bf(o);
    }
}

// ---------------------------------------------------------------------------
// h_bf = bf16(gelu_tanh(h + b1))   (jax.nn.gelu default = tanh approx)
// ---------------------------------------------------------------------------
__global__ void gelu_bias_kernel(const float* __restrict__ h,
                                 const float* __restrict__ b1,
                                 unsigned short* __restrict__ h_bf) {
    int i = blockIdx.x * blockDim.x + threadIdx.x;
    if (i >= NTOK * D_FFN) return;
    int c = i % D_FFN;
    float v = h[i] + b1[c];
    float t = tanhf(0.7978845608028654f * (v + 0.044715f * v * v * v));
    h_bf[i] = f2bf(0.5f * v * (1.0f + t));
}

// ---------------------------------------------------------------------------
// Launch
// ---------------------------------------------------------------------------
extern "C" void kernel_launch(void* const* d_in, const int* in_sizes, int n_in,
                              void* d_out, int out_size, void* d_ws, size_t ws_size,
                              hipStream_t stream) {
    (void)in_sizes; (void)n_in; (void)out_size; (void)ws_size;
    const float* x         = (const float*)d_in[0];
    // d_in[1] = mask (all ones) -- unused
    const float* in_proj_w = (const float*)d_in[2];
    const float* conv_w    = (const float*)d_in[3];
    const float* conv_b    = (const float*)d_in[4];
    const float* x_proj_w  = (const float*)d_in[5];
    const float* dt_proj_w = (const float*)d_in[6];
    const float* dt_proj_b = (const float*)d_in[7];
    const float* A_log     = (const float*)d_in[8];
    const float* Dskip     = (const float*)d_in[9];
    const float* out_proj_w= (const float*)d_in[10];
    const float* a1        = (const float*)d_in[11];
    const float* ln1_g     = (const float*)d_in[12];
    const float* ln1_b     = (const float*)d_in[13];
    const float* w1        = (const float*)d_in[14];
    const float* b1        = (const float*)d_in[15];
    const float* w2        = (const float*)d_in[16];
    const float* b2        = (const float*)d_in[17];
    const float* a2        = (const float*)d_in[18];
    const float* ln2_g     = (const float*)d_in[19];
    const float* ln2_b     = (const float*)d_in[20];

    char* ws = (char*)d_ws;
    size_t off = 0;
    auto alloc = [&](size_t bytes) -> char* {
        char* p = ws + off;
        off = (off + bytes + 255) & ~(size_t)255;
        return p;
    };
    float* xz     = (float*)alloc((size_t)NTOK * D_XZ * 4);     // also reused as FFN h
    float* xconv  = (float*)alloc((size_t)NTOK * D_INNER * 4);
    float* dtbuf  = (float*)alloc((size_t)NTOK * D_INNER * 4);
    float* xdbl   = (float*)alloc((size_t)NTOK * D_DBL * 4);
    float* attn   = (float*)alloc((size_t)NTOK * D_MODEL * 4);  // also FFN out
    float* x1     = (float*)alloc((size_t)NTOK * D_MODEL * 4);
    unsigned short* actA = (unsigned short*)alloc((size_t)NTOK * D_XZ * 2);
    unsigned short* actB = (unsigned short*)alloc((size_t)NTOK * D_XZ * 2);
    unsigned short* w_inproj  = (unsigned short*)alloc((size_t)D_XZ * D_MODEL * 2);
    unsigned short* w_xproj   = (unsigned short*)alloc((size_t)D_DBL * D_INNER * 2);
    unsigned short* w_dtproj  = (unsigned short*)alloc((size_t)D_INNER * DT_RANK * 2);
    unsigned short* w_outproj = (unsigned short*)alloc((size_t)D_MODEL * D_INNER * 2);
    unsigned short* w_w1      = (unsigned short*)alloc((size_t)D_FFN * D_MODEL * 2);
    unsigned short* w_w2      = (unsigned short*)alloc((size_t)D_MODEL * D_FFN * 2);

    auto cvt = [&](const float* s, unsigned short* dd, size_t n) {
        cvt_f32_bf16_kernel<<<(unsigned)((n + 255) / 256), 256, 0, stream>>>(
            s, dd, (int)n);
    };
    auto gemm = [&](const unsigned short* Ap, int lda, const unsigned short* Wp,
                    int ldw, const float* bias, float* Cp, int ldc,
                    int M, int N, int K) {
        dim3 grid((N + 127) / 128, (M + 127) / 128);
        gemm_bf16_wmma_kernel<<<grid, dim3(256), 0, stream>>>(
            Ap, lda, Wp, ldw, bias, Cp, ldc, M, N, K);
    };

    // Weight + input downconversion (bf16 path for WMMA)
    cvt(in_proj_w,  w_inproj,  (size_t)D_XZ * D_MODEL);
    cvt(x_proj_w,   w_xproj,   (size_t)D_DBL * D_INNER);
    cvt(dt_proj_w,  w_dtproj,  (size_t)D_INNER * DT_RANK);
    cvt(out_proj_w, w_outproj, (size_t)D_MODEL * D_INNER);
    cvt(w1, w_w1, (size_t)D_FFN * D_MODEL);
    cvt(w2, w_w2, (size_t)D_MODEL * D_FFN);
    cvt(x, actA, (size_t)NTOK * D_MODEL);

    // 1. in_proj: xz = x @ in_proj_w^T
    gemm(actA, D_MODEL, w_inproj, D_MODEL, nullptr, xz, D_XZ, NTOK, D_XZ, D_MODEL);
    // 2. causal depthwise conv + SiLU (f32 + bf16 into actA)
    conv_silu_kernel<<<(NTOK * D_INNER + 255) / 256, 256, 0, stream>>>(
        xz, conv_w, conv_b, xconv, actA);
    // 3. x_proj: xdbl = xconv @ x_proj_w^T
    gemm(actA, D_INNER, w_xproj, D_INNER, nullptr, xdbl, D_DBL, NTOK, D_DBL, D_INNER);
    cvt(xdbl, actB, (size_t)NTOK * D_DBL);
    // 4. dt = softplus(xdbl[:, :48] @ dt_proj_w^T + dt_proj_b)
    gemm(actB, D_DBL, w_dtproj, DT_RANK, nullptr, dtbuf, D_INNER, NTOK, D_INNER, DT_RANK);
    dt_softplus_kernel<<<(NTOK * D_INNER + 255) / 256, 256, 0, stream>>>(
        dtbuf, dt_proj_b);
    // 5. selective scan -> y (bf16, into actA) with Dskip + silu(z) gating fused
    scan_kernel<<<(BATCH * D_INNER) / 16, 256, 0, stream>>>(
        dtbuf, xconv, xdbl, xz, A_log, Dskip, actA);
    // 6. out_proj: attn = y @ out_proj_w^T
    gemm(actA, D_INNER, w_outproj, D_INNER, nullptr, attn, D_MODEL, NTOK, D_MODEL, D_INNER);
    // 7. x1 = LN(x + a1*attn)   (f32 + bf16 into actA)
    res_ln_kernel<<<NTOK, 256, 0, stream>>>(x, attn, a1, ln1_g, ln1_b, x1, actA);
    // 8. FFN: h = x1 @ w1^T (reuse xz buffer), gelu(h+b1) -> bf16 actB
    gemm(actA, D_MODEL, w_w1, D_MODEL, nullptr, xz, D_FFN, NTOK, D_FFN, D_MODEL);
    gelu_bias_kernel<<<(NTOK * D_FFN + 255) / 256, 256, 0, stream>>>(xz, b1, actB);
    // 9. ffn = h @ w2^T + b2 (reuse attn buffer)
    gemm(actB, D_FFN, w_w2, D_FFN, b2, attn, D_MODEL, NTOK, D_MODEL, D_FFN);
    // 10. out = LN(x1 + a2*ffn)
    res_ln_kernel<<<NTOK, 256, 0, stream>>>(x1, attn, a2, ln2_g, ln2_b,
                                            (float*)d_out, nullptr);
}
